// LevelHDC_65446711657225
// MI455X (gfx1250) — matrix operand order repaired and checked by reading, not compiled, not run on hardware
//
#include <hip/hip_runtime.h>
#include <stdint.h>

// ---------------------------------------------------------------------------
// LevelHDC on MI455X (gfx1250): sparse level-interpolation cast as dense bf16
// GEMM  encoded = A(256x6400) @ G(6400x10000)  on v_wmma_f32_16x16x32_bf16.
//   memory floor : 256 MB of base_hvs @ 23.3 TB/s ~= 11 us (G read exactly once)
//   wmma compute : 32.8 GFLOP bf16                ~= same order
// This revision: K-step 64 (20 wmma per barrier, 100 barriers instead of 200),
// and f32->bf16 packing via v_add + v_perm_b32 (3 VALU/pair instead of ~5) to
// cut issue pressure at 2 waves/SIMD.
// ---------------------------------------------------------------------------

typedef __attribute__((ext_vector_type(16))) __bf16 v16bf;
typedef __attribute__((ext_vector_type(8)))  float  v8f;

#define BATCH     256
#define INPUT_DIM 64
#define LEVELS    100
#define HV_DIM    10000
#define KDIM      (INPUT_DIM * LEVELS)   // 6400
#define KP        (KDIM / 2)             // 3200 packed-u32 per A row
#define NTILE     80                     // output columns per workgroup (125 WGs)
#define LSTRIDE   20                     // u32 per LDS row: 16 K-pairs + pad
                                         //  80B rows: 16B aligned; 16 fragment
                                         //  rows start at banks {20r mod 64}
                                         //  (4-spaced perm) -> conflict-free b128
#define KSTEP     64
#define NITER     (KDIM / KSTEP)         // 100 K-steps

// round-half-up f32->bf16 (ties differ from RNE with probability ~0),
// merged with a single v_perm_b32: 3 VALU per packed pair.
__device__ __forceinline__ uint32_t pack_bf16x2(float lo, float hi) {
  uint32_t a = __builtin_bit_cast(uint32_t, lo) + 0x8000u;
  uint32_t b = __builtin_bit_cast(uint32_t, hi) + 0x8000u;
  // D = {b[31:16], a[31:16]} : sel bytes {7,6,3,2} of {S0=b : S1=a}
  return __builtin_amdgcn_perm(b, a, 0x07060302u);
}

// ---------------------------------------------------------------------------
// Kernel 1: materialize A[b][k] (bf16, packed K-pairs) analytically from x.
// A[b][d*100 + l] = (l==lo)*(1-a) + (l==hi)*a   -- no scatter, no atomics.
// ---------------------------------------------------------------------------
__global__ __launch_bounds__(256)
void build_A(const float* __restrict__ x, uint32_t* __restrict__ A) {
  int idx = blockIdx.x * blockDim.x + threadIdx.x;   // one packed u32 each
  if (idx >= BATCH * KP) return;
  int b = idx / KP;
  int r = idx - b * KP;
  int d = r / (LEVELS / 2);
  int p = r - d * (LEVELS / 2);
  int l0 = 2 * p, l1 = 2 * p + 1;

  float xv = x[b * INPUT_DIM + d];
  float xn = fminf(fmaxf(xv * (float)(LEVELS - 1), 0.0f), (float)(LEVELS - 1));
  int   lo = (int)floorf(xn);
  int   hi = min(lo + 1, LEVELS - 1);
  float a  = xn - (float)lo;

  float w0 = ((l0 == lo) ? (1.0f - a) : 0.0f) + ((l0 == hi) ? a : 0.0f);
  float w1 = ((l1 == lo) ? (1.0f - a) : 0.0f) + ((l1 == hi) ? a : 0.0f);
  A[idx] = pack_bf16x2(w0, w1);
}

// ---------------------------------------------------------------------------
// Kernel 2: encoded(unnormalized) = A @ G.
// 125 workgroups, one 80-column stripe each -> G streamed from HBM once.
// 256 threads = 8 waves; wave w owns rows [32w,32w+32) x 80 cols = 2x5 C tiles.
// Double-buffered LDS, K-step 64 staged as two 32-K chunks; per iteration:
// global loads (next) -> 20 wmma (cur) -> lds stores (next) -> one barrier.
// ---------------------------------------------------------------------------
__global__ __launch_bounds__(256)
void hdc_gemm(const uint32_t* __restrict__ A, const float* __restrict__ G,
              float* __restrict__ out) {
  __shared__ uint32_t lA[2][2][BATCH * LSTRIDE];   // [buf][chunk] 4 x 20 KB
  __shared__ uint32_t lG[2][2][NTILE * LSTRIDE];   // [buf][chunk] 4 x 6.4 KB

  const int t    = threadIdx.x;
  const int lane = t & 31;
  const int wave = t >> 5;        // 0..7
  const int half = lane >> 4;     // 0: lanes 0-15, 1: lanes 16-31
  const int l16  = lane & 15;
  const int n0   = blockIdx.x * NTILE;

  // G staging: 640 float4-tasks per tile (32 K-pairs x 20 col-groups);
  // thread t owns tasks {t, 256+t, 512+t (t<128)}.
  int gkp[3], gn[3];
#pragma unroll
  for (int i = 0; i < 3; ++i) {
    int task = i * 256 + t;
    gkp[i] = task / 20;           // K-pair index 0..31
    gn[i]  = (task % 20) * 4;     // column within stripe
  }
  const bool g2 = (t < 128);

  v8f acc[2][5];
  const v8f vzero = {};
#pragma unroll
  for (int mt = 0; mt < 2; ++mt)
#pragma unroll
    for (int nt = 0; nt < 5; ++nt) acc[mt][nt] = vzero;

  uint4  ra[8];                 // A staging registers (8 x b128)
  float4 rgA[3], rgB[3];        // G staging: [task]{row k, row k+1}

  auto load_tile = [&](int kb) {
    const uint32_t* Ab = A + (kb >> 1);
#pragma unroll
    for (int j = 0; j < 8; ++j) {         // 2048 b128-tasks / 256 threads
      int task = j * 256 + t;
      int m    = task >> 3;               // row 0..255
      int g    = task & 7;                // pair-group 0..7 (32 pairs)
      ra[j] = *(const uint4*)(Ab + (size_t)m * KP + g * 4);
    }
#pragma unroll
    for (int i = 0; i < 2; ++i) {
      const float* p = G + (size_t)(kb + 2 * gkp[i]) * HV_DIM + (n0 + gn[i]);
      rgA[i] = *(const float4*)p;
      rgB[i] = *(const float4*)(p + HV_DIM);
    }
    if (g2) {
      const float* p = G + (size_t)(kb + 2 * gkp[2]) * HV_DIM + (n0 + gn[2]);
      rgA[2] = *(const float4*)p;
      rgB[2] = *(const float4*)(p + HV_DIM);
    }
  };

  auto store_g_task = [&](int buf, int i) {
    int c = gkp[i] >> 4, kpc = gkp[i] & 15;
    uint32_t* bG = lG[buf][c];
    bG[(gn[i] + 0) * LSTRIDE + kpc] = pack_bf16x2(rgA[i].x, rgB[i].x);
    bG[(gn[i] + 1) * LSTRIDE + kpc] = pack_bf16x2(rgA[i].y, rgB[i].y);
    bG[(gn[i] + 2) * LSTRIDE + kpc] = pack_bf16x2(rgA[i].z, rgB[i].z);
    bG[(gn[i] + 3) * LSTRIDE + kpc] = pack_bf16x2(rgA[i].w, rgB[i].w);
  };

  auto store_tile = [&](int buf) {
#pragma unroll
    for (int j = 0; j < 8; ++j) {
      int task = j * 256 + t;
      int m    = task >> 3;
      int g    = task & 7;
      int c    = g >> 2;                  // chunk 0: pairs 0..15, chunk 1: 16..31
      int gg   = g & 3;
      *(uint4*)(lA[buf][c] + m * LSTRIDE + gg * 4) = ra[j];   // 16B-aligned b128
    }
    store_g_task(buf, 0);
    store_g_task(buf, 1);
    if (g2) store_g_task(buf, 2);
  };

  auto compute_tile = [&](int buf) {
    union ABu { uint32_t u[8]; v16bf v; };
#pragma unroll
    for (int c = 0; c < 2; ++c) {
      const uint32_t* bA = lA[buf][c];
      const uint32_t* bG = lG[buf][c];
      // A 16x32 fragment (ISA 7.12.2): lane<16 row M=l16 holds K-pairs
      // {0..3,8..11}; lanes>=16 hold {4..7,12..15}. Two b128 loads each.
      ABu af[2];
#pragma unroll
      for (int mt = 0; mt < 2; ++mt) {
        const uint32_t* base = bA + (wave * 32 + mt * 16 + l16) * LSTRIDE;
#pragma unroll
        for (int q = 0; q < 4; ++q) {
          af[mt].u[q]     = base[half * 4 + q];
          af[mt].u[4 + q] = base[8 + half * 4 + q];
        }
      }
      // B 32x16 fragment (same K order as A): lane<16 K-pairs {0..3,8..11}; lanes>=16 {4..7,12..15}.
      ABu bfr[5];
#pragma unroll
      for (int nt = 0; nt < 5; ++nt) {
        const uint32_t* base = bG + (nt * 16 + l16) * LSTRIDE;
#pragma unroll
        for (int q = 0; q < 8; ++q) bfr[nt].u[q] = base[(q >> 2) * 8 + half * 4 + (q & 3)];
      }
#pragma unroll
      for (int nt = 0; nt < 5; ++nt)
#pragma unroll
        for (int mt = 0; mt < 2; ++mt)
          acc[mt][nt] = __builtin_amdgcn_wmma_f32_16x16x32_bf16(
              false, af[mt].v, false, bfr[nt].v,
              (short)0, acc[mt][nt], false, false);
    }
  };

  // ---- software pipeline: prologue fills buffer 0
  load_tile(0);
  store_tile(0);
  __syncthreads();

  for (int it = 0; it < NITER; ++it) {
    const int buf = it & 1;
    const bool have_next = (it + 1) < NITER;
    if (have_next) {
      load_tile((it + 1) * KSTEP);
      if ((it + 2) < NITER) {             // warm L2 two tiles ahead (G stream)
        const int kb2 = (it + 2) * KSTEP;
#pragma unroll
        for (int i = 0; i < 2; ++i) {
          const float* p = G + (size_t)(kb2 + 2 * gkp[i]) * HV_DIM + (n0 + gn[i]);
          __builtin_prefetch(p, 0, 1);
          __builtin_prefetch(p + HV_DIM, 0, 1);
        }
      }
    }
    compute_tile(buf);
    if (have_next) store_tile(buf ^ 1);
    __syncthreads();
  }

  // ---- epilogue: C layout: VGPR r, lanes<16 -> M=r, lanes>=16 -> M=r+8
#pragma unroll
  for (int mt = 0; mt < 2; ++mt) {
#pragma unroll
    for (int nt = 0; nt < 5; ++nt) {
      int col = n0 + nt * 16 + l16;
#pragma unroll
      for (int r = 0; r < 8; ++r) {
        int row = wave * 32 + mt * 16 + r + half * 8;
        out[(size_t)row * HV_DIM + col] = acc[mt][nt][r];
      }
    }
  }
}

// ---------------------------------------------------------------------------
// Kernel 3: deterministic per-row L2 normalization (1 block per batch row).
// ---------------------------------------------------------------------------
__global__ __launch_bounds__(256)
void row_normalize(float* __restrict__ out) {
  __shared__ float red[256];
  const int b = blockIdx.x;
  float* row = out + (size_t)b * HV_DIM;

  float s = 0.0f;
  for (int h = threadIdx.x; h < HV_DIM; h += 256) {
    float v = row[h];
    s += v * v;
  }
  red[threadIdx.x] = s;
  __syncthreads();
  for (int off = 128; off > 0; off >>= 1) {
    if (threadIdx.x < off) red[threadIdx.x] += red[threadIdx.x + off];
    __syncthreads();
  }
  float total = red[0];
  float inv   = rsqrtf(total);
  inv = inv * (1.5f - 0.5f * total * inv * inv);   // one Newton step
  for (int h = threadIdx.x; h < HV_DIM; h += 256) row[h] *= inv;
}

// ---------------------------------------------------------------------------
extern "C" void kernel_launch(void* const* d_in, const int* in_sizes, int n_in,
                              void* d_out, int out_size, void* d_ws, size_t ws_size,
                              hipStream_t stream) {
  const float* x    = (const float*)d_in[0];   // (256, 64) f32
  const float* base = (const float*)d_in[1];   // (64, 100, 10000) f32 == G(6400,10000)
  float* out        = (float*)d_out;           // (256, 10000) f32
  uint32_t* A       = (uint32_t*)d_ws;         // 256 x 3200 packed bf16 pairs (3.3 MB)

  build_A<<<(BATCH * KP + 255) / 256, 256, 0, stream>>>(x, A);
  hdc_gemm<<<HV_DIM / NTILE, 256, 0, stream>>>(A, base, out);   // 125 workgroups
  row_normalize<<<BATCH, 256, 0, stream>>>(out);
}
